// NeRFNetwork_2259152798107
// MI455X (gfx1250) — matrix operand order repaired and checked by reading, not compiled, not run on hardware
//
#include <hip/hip_runtime.h>
#include <hip/hip_bf16.h>
#include <math.h>

typedef __attribute__((ext_vector_type(16))) _Float16 v16h;
typedef __attribute__((ext_vector_type(8)))  _Float16 v8h;
typedef __attribute__((ext_vector_type(2)))  _Float16 v2h;
typedef __attribute__((ext_vector_type(8)))  float    v8f;

#define ACT_W 96          // per-point activation row width (halves) in LDS
#define TLOG 19
#define TSIZE (1u << TLOG)

__device__ const float RES_TAB[16] = {
    16.f, 20.f, 25.f, 32.f, 40.f, 50.f, 64.f, 80.f,
    101.f, 128.f, 161.f, 203.f, 256.f, 322.f, 406.f, 512.f};

__device__ __forceinline__ v8f WMMA(v16h a, v16h b, v8f c) {
  // D = A(16x32 f16) * B(32x16 f16) + C(16x16 f32)
  return __builtin_amdgcn_wmma_f32_16x16x32_f16(
      false, a, false, b, (short)0, c, false, false);
}

union Frag {
  v16h v;
  v8h h[2];
};

// A fragment (16-bit A 16x32): lane holds row m = lane&15; halves j=0..15 are
// K = colbase + g*16 + (0..15).  Both pieces
// are 16B-aligned contiguous LDS -> two ds_load_b128.
__device__ __forceinline__ v16h load_a(const _Float16* act, int row, int g,
                                       int colbase) {
  const v8h* p = (const v8h*)(act + row * ACT_W + colbase + g * 16);
  Frag u;
  u.h[0] = p[0];
  u.h[1] = p[1];  // +8 halves
  return u.v;
}

// Weight fragment index: element (k,n) of a Kx(16*ntiles) matrix goes to
// ((chunk*ntiles + tile)*32 + lane)*16 + j so that each lane's B fragment
// is 16 contiguous halves (32 bytes) -> two ds_load_b128.
__device__ __forceinline__ int frag_idx(int k, int n, int ntiles) {
  int tile = n >> 4;
  int lane = (((k >> 4) & 1) << 4) | (n & 15);
  int chunk = k >> 5;
  int j = k & 15;
  return ((chunk * ntiles + tile) * 32 + lane) * 16 + j;
}

__device__ __forceinline__ v16h load_b(const _Float16* Wf, int ntiles,
                                       int chunk, int tile, int lane) {
  const v8h* p =
      (const v8h*)(Wf + (((chunk * ntiles + tile) * 32 + lane) << 4));
  Frag u;
  u.h[0] = p[0];
  u.h[1] = p[1];
  return u.v;
}

// 64 -> 64 fused linear + bias + relu, in-place on per-wave act tile.
__device__ __forceinline__ void layer64(_Float16* act, const _Float16* Wf,
                                        const float* bias, int lane) {
  int g = lane >> 4;
  int row = lane & 15;
  int ncol = lane & 15;
  v16h a0 = load_a(act, row, g, 0);
  v16h a1 = load_a(act, row, g, 32);
  __syncthreads();  // all A fragments in regs before overwriting tiles
#pragma unroll
  for (int nt = 0; nt < 4; ++nt) {
    v8f d = {};
    d = WMMA(a0, load_b(Wf, 4, 0, nt, lane), d);
    d = WMMA(a1, load_b(Wf, 4, 1, nt, lane), d);
    float bv = bias[nt * 16 + ncol];
#pragma unroll
    for (int j = 0; j < 8; ++j)
      act[(8 * g + j) * ACT_W + nt * 16 + ncol] =
          (_Float16)fmaxf(d[j] + bv, 0.f);
  }
  __syncthreads();
}

__global__ __launch_bounds__(128) void nerf_fused(
    const float* __restrict__ coords, const float* __restrict__ dirs,
    const float* __restrict__ emb, const float* __restrict__ dw1,
    const float* __restrict__ db1, const float* __restrict__ dw2,
    const float* __restrict__ db2, const float* __restrict__ cw1,
    const float* __restrict__ cb1, const float* __restrict__ cw2,
    const float* __restrict__ cb2, const float* __restrict__ cw3,
    const float* __restrict__ cb3, const float* __restrict__ cw4,
    const float* __restrict__ cb4, float* __restrict__ out_density,
    float* __restrict__ out_color) {
  // all weights stored PRE-SWIZZLED in fragment order (see frag_idx)
  __shared__ __align__(16) _Float16 s_dw1[32 * 64];
  __shared__ __align__(16) _Float16 s_dw2[64 * 16];
  __shared__ __align__(16) _Float16 s_cw1[32 * 64];
  __shared__ __align__(16) _Float16 s_cw2[64 * 64];
  __shared__ __align__(16) _Float16 s_cw3[64 * 64];
  __shared__ __align__(16) _Float16 s_cw4[64 * 16];  // 64x3 zero-padded
  __shared__ float s_db1[64], s_db2[16], s_cb1[64], s_cb2[64], s_cb3[64],
      s_cb4[16];
  __shared__ __align__(16) _Float16 s_act[4 * 16 * ACT_W];

  int tid = threadIdx.x;
  // ---- stage weights (f32 -> f16, swizzled to fragment order) ----
  for (int i = tid; i < 32 * 64; i += 128)
    s_dw1[frag_idx(i >> 6, i & 63, 4)] = (_Float16)dw1[i];
  for (int i = tid; i < 64 * 16; i += 128)
    s_dw2[frag_idx(i >> 4, i & 15, 1)] = (_Float16)dw2[i];
  for (int i = tid; i < 32 * 64; i += 128)
    s_cw1[frag_idx(i >> 6, i & 63, 4)] = (_Float16)cw1[i];
  for (int i = tid; i < 64 * 64; i += 128)
    s_cw2[frag_idx(i >> 6, i & 63, 4)] = (_Float16)cw2[i];
  for (int i = tid; i < 64 * 64; i += 128)
    s_cw3[frag_idx(i >> 6, i & 63, 4)] = (_Float16)cw3[i];
  for (int i = tid; i < 64 * 16; i += 128) {
    int k = i >> 4, n = i & 15;
    s_cw4[frag_idx(k, n, 1)] = (_Float16)(n < 3 ? cw4[k * 3 + n] : 0.0f);
  }
  if (tid < 64) {
    s_db1[tid] = db1[tid];
    s_cb1[tid] = cb1[tid];
    s_cb2[tid] = cb2[tid];
    s_cb3[tid] = cb3[tid];
  }
  if (tid < 16) {
    s_db2[tid] = db2[tid];
    s_cb4[tid] = (tid < 3) ? cb4[tid] : 0.0f;
  }
  __syncthreads();

  int lane = tid & 31;
  int wv = tid >> 5;
  int g = lane >> 4;
  int row = lane & 15;   // A-fragment row
  int ncol = lane & 15;  // C/D-fragment column
  _Float16* act = s_act + wv * 16 * ACT_W;
  int pbase = (blockIdx.x * 4 + wv) * 16;

  // ---- phase 1: multiresolution hash encoding ----
  // lane = (psub, point): each lane owns one point (coords hoisted) and
  // covers 8 of 16 levels; lane+16 covers the other 8.
  {
    int pl = lane & 15;
    int psub = lane >> 4;
    int p = pbase + pl;
    float cx = coords[p * 3 + 0];
    float cy = coords[p * 3 + 1];
    float cz = coords[p * 3 + 2];
    _Float16* frow = act + pl * ACT_W + 64;
#pragma unroll 2
    for (int it = 0; it < 8; ++it) {
      int lvl = it * 2 + psub;
      float r = RES_TAB[lvl];
      const float2* embL = (const float2*)emb + (size_t)lvl * TSIZE;
      float px = cx * r, py = cy * r, pz = cz * r;
      float fx = floorf(px), fy = floorf(py), fz = floorf(pz);
      unsigned x0 = (unsigned)fx, y0 = (unsigned)fy, z0 = (unsigned)fz;
      float wx[2], wy[2], wz[2];
      wx[1] = px - fx; wx[0] = 1.f - wx[1];
      wy[1] = py - fy; wy[0] = 1.f - wy[1];
      wz[1] = pz - fz; wz[0] = 1.f - wz[1];
      unsigned hx[2] = {x0, x0 + 1u};
      unsigned hy[2] = {y0 * 2654435761u, (y0 + 1u) * 2654435761u};
      unsigned hz[2] = {z0 * 805459861u, (z0 + 1u) * 805459861u};
      float f0 = 0.f, f1 = 0.f;
#pragma unroll
      for (int c = 0; c < 8; ++c) {
        unsigned idx = (hx[c & 1] ^ hy[(c >> 1) & 1] ^ hz[c >> 2]) & (TSIZE - 1u);
        float2 e = embL[idx];
        float wgt = wx[c & 1] * wy[(c >> 1) & 1] * wz[c >> 2];
        f0 += wgt * e.x;
        f1 += wgt * e.y;
      }
      v2h pk = {(_Float16)f0, (_Float16)f1};
      *(v2h*)(frow + 2 * lvl) = pk;  // one ds_store_b32
    }
  }
  __syncthreads();

  // ---- phase 2: density layer 1: h = relu(feat @ dw1 + db1) ----
  {
    v16h a = load_a(act, row, g, 64);  // feat is 16x32 at cols 64..95
#pragma unroll
    for (int nt = 0; nt < 4; ++nt) {
      v8f d = {};
      d = WMMA(a, load_b(s_dw1, 4, 0, nt, lane), d);
      float bv = s_db1[nt * 16 + ncol];
#pragma unroll
      for (int j = 0; j < 8; ++j)
        act[(8 * g + j) * ACT_W + nt * 16 + ncol] =
            (_Float16)fmaxf(d[j] + bv, 0.f);
    }
  }
  __syncthreads();

  // ---- phase 3: density_out = h @ dw2 + db2 ; density = exp(col0);
  //      assemble color input [sh | density_out] at cols 64..95 ----
  {
    v16h a0 = load_a(act, row, g, 0);
    v16h a1 = load_a(act, row, g, 32);
    v8f d = {};
    d = WMMA(a0, load_b(s_dw2, 1, 0, 0, lane), d);
    d = WMMA(a1, load_b(s_dw2, 1, 1, 0, lane), d);
    float bv = s_db2[ncol];
#pragma unroll
    for (int j = 0; j < 8; ++j) {
      float v = d[j] + bv;
      int m = 8 * g + j;
      act[m * ACT_W + 80 + ncol] = (_Float16)v;
      if (ncol == 0) out_density[pbase + m] = expf(v);
    }
    if (lane < 16) {  // SH deg-4 of normalized direction, one point per lane
      int p = pbase + lane;
      float dx = dirs[p * 3 + 0], dy = dirs[p * 3 + 1], dz = dirs[p * 3 + 2];
      float inv = rsqrtf(dx * dx + dy * dy + dz * dz);
      float x = dx * inv, y = dy * inv, z = dz * inv;
      float xx = x * x, yy = y * y, zz = z * z;
      float xy = x * y, yz = y * z, xz = x * z;
      float sh[16];
      sh[0] = 0.28209479177387814f;
      sh[1] = -0.48860251190291987f * y;
      sh[2] = 0.48860251190291987f * z;
      sh[3] = -0.48860251190291987f * x;
      sh[4] = 1.0925484305920792f * xy;
      sh[5] = -1.0925484305920792f * yz;
      sh[6] = 0.94617469575756f * zz - 0.31539156525252f;
      sh[7] = -1.0925484305920792f * xz;
      sh[8] = 0.5462742152960396f * (xx - yy);
      sh[9] = 0.5900435899266435f * y * (-3.0f * xx + yy);
      sh[10] = 2.890611442640554f * xy * z;
      sh[11] = 0.4570457994644657f * y * (1.0f - 5.0f * zz);
      sh[12] = 0.3731763325901154f * z * (5.0f * zz - 3.0f);
      sh[13] = 0.4570457994644657f * x * (1.0f - 5.0f * zz);
      sh[14] = 1.445305721320277f * z * (xx - yy);
      sh[15] = 0.5900435899266435f * x * (xx - 3.0f * yy);
      v8h s0, s1;
#pragma unroll
      for (int k = 0; k < 8; ++k) {
        s0[k] = (_Float16)sh[k];
        s1[k] = (_Float16)sh[k + 8];
      }
      v8h* rowp = (v8h*)(act + lane * ACT_W + 64);
      rowp[0] = s0;  // two ds_store_b128
      rowp[1] = s1;
    }
  }
  __syncthreads();

  // ---- phase 4: color layer 1: x = relu(inp @ cw1 + cb1) ----
  {
    v16h a = load_a(act, row, g, 64);  // inp 16x32 at cols 64..95
#pragma unroll
    for (int nt = 0; nt < 4; ++nt) {
      v8f d = {};
      d = WMMA(a, load_b(s_cw1, 4, 0, nt, lane), d);
      float bv = s_cb1[nt * 16 + ncol];
#pragma unroll
      for (int j = 0; j < 8; ++j)
        act[(8 * g + j) * ACT_W + nt * 16 + ncol] =
            (_Float16)fmaxf(d[j] + bv, 0.f);
    }
  }
  __syncthreads();

  // ---- phases 5,6: two 64->64 relu layers ----
  layer64(act, s_cw2, s_cb2, lane);
  layer64(act, s_cw3, s_cb3, lane);

  // ---- phase 7: color = sigmoid(x @ cw4 + cb4), cols 0..2 ----
  {
    v16h a0 = load_a(act, row, g, 0);
    v16h a1 = load_a(act, row, g, 32);
    v8f d = {};
    d = WMMA(a0, load_b(s_cw4, 1, 0, 0, lane), d);
    d = WMMA(a1, load_b(s_cw4, 1, 1, 0, lane), d);
    if (ncol < 3) {
      float bv = s_cb4[ncol];
#pragma unroll
      for (int j = 0; j < 8; ++j) {
        float v = d[j] + bv;
        v = 1.f / (1.f + expf(-v));
        out_color[(size_t)(pbase + 8 * g + j) * 3 + ncol] = v;
      }
    }
  }
}

extern "C" void kernel_launch(void* const* d_in, const int* in_sizes, int n_in,
                              void* d_out, int out_size, void* d_ws,
                              size_t ws_size, hipStream_t stream) {
  const float* coords = (const float*)d_in[0];
  const float* dirs = (const float*)d_in[1];
  const float* emb = (const float*)d_in[2];
  const float* dw1 = (const float*)d_in[3];
  const float* db1 = (const float*)d_in[4];
  const float* dw2 = (const float*)d_in[5];
  const float* db2 = (const float*)d_in[6];
  const float* cw1 = (const float*)d_in[7];
  const float* cb1 = (const float*)d_in[8];
  const float* cw2 = (const float*)d_in[9];
  const float* cb2 = (const float*)d_in[10];
  const float* cw3 = (const float*)d_in[11];
  const float* cb3 = (const float*)d_in[12];
  const float* cw4 = (const float*)d_in[13];
  const float* cb4 = (const float*)d_in[14];

  int npts = in_sizes[0] / 3;  // B = 262144
  float* out_density = (float*)d_out;
  float* out_color = (float*)d_out + npts;

  int nblocks = npts / 64;  // 4 waves/block x 16 points/wave
  nerf_fused<<<nblocks, 128, 0, stream>>>(
      coords, dirs, emb, dw1, db1, dw2, db2, cw1, cb1, cw2, cb2, cw3, cb3,
      cw4, cb4, out_density, out_color);
}
